// BindNode23SAGEConv_62715112456263
// MI455X (gfx1250) — matrix operand order repaired and checked by reading, not compiled, not run on hardware
//
#include <hip/hip_runtime.h>

// ---------- WMMA types (CDNA5 / gfx1250, wave32) ----------
typedef __attribute__((ext_vector_type(16))) __bf16 v16bf;
typedef __attribute__((ext_vector_type(8)))  float  v8f;

union BF16Frag { v16bf v; __bf16 e[16]; };

// ---------- Weight pre-pack into WMMA B-fragment layout ----------
// B (32x16 bf16 per k-tile) lane layout: lane l, h = l>>4, col = l&15:
//   vector elem j (j=0..15)  ->  B[k = kt*32 + (j<8 ? h*8 + j : 16 + h*8 + j-8)][col]
// Columns [0, halfCols) come from Wl, [halfCols, 2*halfCols) from Wr
// (fused dual projection). K zero-padded past Krows.
__global__ void pack_weights_kernel(const float* __restrict__ Wl,
                                    const float* __restrict__ Wr,
                                    int halfCols, int Krows, int Ktiles, int NT,
                                    __bf16* __restrict__ P) {
  int tid = blockIdx.x * blockDim.x + threadIdx.x;
  int total = Ktiles * NT * 32;
  if (tid >= total) return;
  int lane = tid & 31;
  int nt   = (tid >> 5) % NT;
  int kt   = (tid >> 5) / NT;
  int h    = lane >> 4;
  int col  = (lane & 15) + nt * 16;
  const float* W = (col < halfCols) ? Wl : Wr;
  int c = (col < halfCols) ? col : (col - halfCols);
  __bf16* out = P + (size_t)tid * 16;
#pragma unroll
  for (int j = 0; j < 16; ++j) {
    int k = kt * 32 + (j < 8 ? h * 8 + j : 16 + h * 8 + (j - 8));
    float v = (k < Krows) ? W[(size_t)k * halfCols + c] : 0.0f;
    out[j] = (__bf16)v;
  }
}

// Build the 16x32 bf16 A fragment from 16 f32 values (native cvt -> pk_bf16).
__device__ __forceinline__ v16bf make_a_frag(float4 f0, float4 f1, float4 g0,
                                             float4 g1) {
  BF16Frag a;
  a.e[0]  = (__bf16)f0.x; a.e[1]  = (__bf16)f0.y;
  a.e[2]  = (__bf16)f0.z; a.e[3]  = (__bf16)f0.w;
  a.e[4]  = (__bf16)f1.x; a.e[5]  = (__bf16)f1.y;
  a.e[6]  = (__bf16)f1.z; a.e[7]  = (__bf16)f1.w;
  a.e[8]  = (__bf16)g0.x; a.e[9]  = (__bf16)g0.y;
  a.e[10] = (__bf16)g0.z; a.e[11] = (__bf16)g0.w;
  a.e[12] = (__bf16)g1.x; a.e[13] = (__bf16)g1.y;
  a.e[14] = (__bf16)g1.z; a.e[15] = (__bf16)g1.w;
  return a.v;
}

// ---------- Tiled GEMM: C[M x (NT*16)] = A[M x K] * Bpacked ----------
// One wave per workgroup; each wave owns one 16-row M tile and computes all
// NT n-tiles, so each A element is loaded/converted exactly once.
// A fragment (16x32 bf16): lane half h holds K = {h*8..h*8+7, 16+h*8..+7}.
template <int NT>
__global__ void __launch_bounds__(32)
wmma_gemm_kernel(const float* __restrict__ A, int K,
                 const __bf16* __restrict__ Bp,
                 float* __restrict__ C, int ldc, int M) {
  const int lane = threadIdx.x & 31;
  const int h    = lane >> 4;
  const int col  = lane & 15;
  const int mtile = blockIdx.x;

  int arowIdx = mtile * 16 + col;
  if (arowIdx >= M) arowIdx = M - 1;           // uniform-EXEC safe clamp
  const float* arow = A + (size_t)arowIdx * K;

  v8f acc[NT];
#pragma unroll
  for (int t = 0; t < NT; ++t)
#pragma unroll
    for (int i = 0; i < 8; ++i) acc[t][i] = 0.0f;

  const int kmain = K >> 5;                    // full 32-wide K tiles

  for (int kt = 0; kt < kmain; ++kt) {
    const int k0 = kt * 32;
    __builtin_prefetch(arow + k0 + 64, 0, 1);  // 2 K-tiles ahead
    const float4 f0 = *(const float4*)(arow + k0 + h * 8);
    const float4 f1 = *(const float4*)(arow + k0 + h * 8 + 4);
    const float4 g0 = *(const float4*)(arow + k0 + 16 + h * 8);
    const float4 g1 = *(const float4*)(arow + k0 + 16 + h * 8 + 4);
    const v16bf a = make_a_frag(f0, f1, g0, g1);
#pragma unroll
    for (int nt = 0; nt < NT; ++nt) {
      BF16Frag b;
      b.v = *(const v16bf*)(Bp + ((size_t)(kt * NT + nt) * 32 + lane) * 16);
      acc[nt] = __builtin_amdgcn_wmma_f32_16x16x32_bf16(
          false, a, false, b.v, (short)0, acc[nt], false, false);
    }
  }

  if (K & 31) {                                // single guarded epilogue tile
    const int k0 = kmain * 32;
    float f[16];
#pragma unroll
    for (int i = 0; i < 8; ++i) {
      int ka = k0 + h * 8 + i;
      int kb = k0 + 16 + h * 8 + i;
      f[i]     = (ka < K) ? arow[ka] : 0.0f;
      f[8 + i] = (kb < K) ? arow[kb] : 0.0f;
    }
    BF16Frag a;
#pragma unroll
    for (int i = 0; i < 16; ++i) a.e[i] = (__bf16)f[i];
#pragma unroll
    for (int nt = 0; nt < NT; ++nt) {
      BF16Frag b;
      b.v = *(const v16bf*)(Bp + ((size_t)(kmain * NT + nt) * 32 + lane) * 16);
      acc[nt] = __builtin_amdgcn_wmma_f32_16x16x32_bf16(
          false, a.v, false, b.v, (short)0, acc[nt], false, false);
    }
  }

  // C/D layout: VGPR r, lane l -> M = (l>>4)*8 + r, N = l&15
  const int rbase = mtile * 16 + h * 8;
#pragma unroll
  for (int nt = 0; nt < NT; ++nt) {
    const int cc = nt * 16 + col;
#pragma unroll
    for (int r = 0; r < 8; ++r) {
      const int rr = rbase + r;
      if (rr < M) C[(size_t)rr * ldc + cc] = acc[nt][r];
    }
  }
}

// ---------- Graph aggregation ----------
__global__ void deg_kernel(const int* __restrict__ dst, int nE,
                           float* __restrict__ deg) {
  int e = blockIdx.x * blockDim.x + threadIdx.x;
  if (e < nE) atomicAdd(&deg[dst[e]], 1.0f);
}

__global__ void scatter_add_kernel(const float* __restrict__ H, int ldH, int colOff,
                                   const int* __restrict__ src,
                                   const int* __restrict__ dst,
                                   float* __restrict__ acc, int ncols, int nE) {
  int e = blockIdx.x;
  if (e >= nE) return;
  int c = threadIdx.x;
  int s = src[e], d = dst[e];
  atomicAdd(&acc[(size_t)d * ncols + c], H[(size_t)s * ldH + colOff + c]);
}

// out = acc/max(deg,1) + b + H[:, colOff:colOff+ncols]; optional ReLU
template <bool RELU>
__global__ void finish_kernel(const float* __restrict__ acc,
                              const float* __restrict__ deg,
                              const float* __restrict__ b,
                              const float* __restrict__ H, int ldH, int colOff,
                              float* __restrict__ out, int ncols, int n) {
  int i = blockIdx.x * blockDim.x + threadIdx.x;
  if (i >= n * ncols) return;
  int node = i / ncols, c = i % ncols;
  float dg = deg[node];
  dg = dg > 1.0f ? dg : 1.0f;
  float v = acc[i] / dg + b[c] + H[(size_t)node * ldH + colOff + c];
  if (RELU) v = v > 0.0f ? v : 0.0f;
  out[i] = v;
}

// ---------- Launch ----------
extern "C" void kernel_launch(void* const* d_in, const int* in_sizes, int n_in,
                              void* d_out, int out_size, void* d_ws, size_t ws_size,
                              hipStream_t stream) {
  const float* features = (const float*)d_in[0];
  const int*   edges    = (const int*)d_in[1];
  // d_in[2]=edges2, d_in[3]=edge_features: unused by the model
  const float* W1_l = (const float*)d_in[4];
  const float* b1_l = (const float*)d_in[5];
  const float* W1_r = (const float*)d_in[6];
  const float* W2_l = (const float*)d_in[7];
  const float* b2_l = (const float*)d_in[8];
  const float* W2_r = (const float*)d_in[9];

  const int C_IN = 1044, C_HID = 128, C_OUT = 64;
  const int nN = in_sizes[0] / C_IN;   // 70000
  const int nE = in_sizes[1] / 2;      // 500000
  const int* src = edges;              // edges[0, :]
  const int* dst = edges + nE;         // edges[1, :]

  const int KT1 = (C_IN + 31) / 32;    // 33 (zero-padded K)
  const int NT1 = (2 * C_HID) / 16;    // 16 n-tiles (fused [W1_l|W1_r])
  const int KT2 = (C_HID + 31) / 32;   // 4
  const int NT2 = (2 * C_OUT) / 16;    // 8  (fused [W2_l|W2_r])

  // ---- workspace carve-out (256B aligned) ----
  char* ws = (char*)d_ws;
  size_t off = 0;
  auto carve = [&](size_t bytes) -> char* {
    char* p = ws + off;
    off = (off + bytes + 255) & ~(size_t)255;
    return p;
  };
  __bf16* P1  = (__bf16*)carve((size_t)KT1 * NT1 * 32 * 16 * 2);
  __bf16* P2  = (__bf16*)carve((size_t)KT2 * NT2 * 32 * 16 * 2);
  float* H1   = (float*)carve((size_t)nN * 256 * 4);  // [X@W1_l | X@W1_r]
  float* x1   = (float*)carve((size_t)nN * 128 * 4);  // layer-1 output
  float* acc1 = (float*)carve((size_t)nN * 128 * 4);  // segment sum L1; reused as H2
  float* acc2 = (float*)carve((size_t)nN * 64 * 4);   // segment sum L2
  float* deg  = (float*)carve((size_t)nN * 4);
  float* H2   = acc1;  // acc1 dead after finish1 -> alias with [x1@W2_l | x1@W2_r]

  // re-zero accumulators every call (graph replay must be state-free)
  hipMemsetAsync(acc1, 0, (size_t)nN * 128 * 4, stream);
  hipMemsetAsync(acc2, 0, (size_t)nN * 64 * 4, stream);
  hipMemsetAsync(deg,  0, (size_t)nN * 4, stream);

  // pack weights into WMMA B-fragment layout (tiny; per call for determinism)
  {
    int tot1 = KT1 * NT1 * 32;
    pack_weights_kernel<<<(tot1 + 255) / 256, 256, 0, stream>>>(
        W1_l, W1_r, C_HID, C_IN, KT1, NT1, P1);
    int tot2 = KT2 * NT2 * 32;
    pack_weights_kernel<<<(tot2 + 255) / 256, 256, 0, stream>>>(
        W2_l, W2_r, C_OUT, C_HID, KT2, NT2, P2);
  }

  deg_kernel<<<(nE + 255) / 256, 256, 0, stream>>>(dst, nE, deg);

  const int mtiles = (nN + 15) / 16;   // 4375; one wave-sized workgroup each

  // Layer 1: fused dual projection (reads 292MB of X exactly once)
  wmma_gemm_kernel<16><<<mtiles, 32, 0, stream>>>(
      features, C_IN, P1, H1, 2 * C_HID, nN);
  scatter_add_kernel<<<nE, C_HID, 0, stream>>>(H1, 2 * C_HID, 0, src, dst,
                                               acc1, C_HID, nE);
  finish_kernel<true><<<((size_t)nN * C_HID + 255) / 256, 256, 0, stream>>>(
      acc1, deg, b1_l, H1, 2 * C_HID, C_HID, x1, C_HID, nN);

  // Layer 2
  wmma_gemm_kernel<8><<<mtiles, 32, 0, stream>>>(
      x1, C_HID, P2, H2, 2 * C_OUT, nN);
  scatter_add_kernel<<<nE, C_OUT, 0, stream>>>(H2, 2 * C_OUT, 0, src, dst,
                                               acc2, C_OUT, nE);
  finish_kernel<false><<<((size_t)nN * C_OUT + 255) / 256, 256, 0, stream>>>(
      acc2, deg, b2_l, H2, 2 * C_OUT, C_OUT, (float*)d_out, C_OUT, nN);
}
